// PlaceRecognitionGCN_43430709297955
// MI455X (gfx1250) — matrix operand order repaired and checked by reading, not compiled, not run on hardware
//
#include <hip/hip_runtime.h>
#include <hip/hip_bf16.h>
#include <math.h>

// ---------------------------------------------------------------------------
// PlaceRecognitionGCN for MI455X (gfx1250, wave32).
//
// Roofline: edge gather/scatter (~1.5 GB) dominates at 23.3 TB/s (~65us).
// GEMMs (16.8 + 8.4 GFLOP) are done with v_wmma_f32_16x16x32_bf16 so they
// stay memory-bound (~15us of HBM traffic) instead of f32-VALU bound.
// fp32 accumulation, bf16 operands (same exponent range as fp32).
// ---------------------------------------------------------------------------

typedef __attribute__((ext_vector_type(16))) __bf16 v16bf;
typedef __attribute__((ext_vector_type(8)))  __bf16 v8bf;
typedef __attribute__((ext_vector_type(8)))  float  v8f;

#define GID (blockIdx.x * blockDim.x + threadIdx.x)

// ------------------------------- degree ------------------------------------
__global__ void gcn_fill_one(float* p, int n) {
  int i = GID; if (i < n) p[i] = 1.0f;            // self-loop contributes 1
}
__global__ void gcn_deg_edge(const int* __restrict__ dst, float* deg, int E) {
  int e = GID; if (e < E) unsafeAtomicAdd(&deg[dst[e]], 1.0f);
}
__global__ void gcn_deg_fin(float* p, int n) {
  int i = GID; if (i < n) p[i] = rsqrtf(p[i]);    // deg >= 1 always
}

// ------------------------------- GEMM (WMMA) --------------------------------
// C[M,N] = A[M,K] * W[K,N], fp32 in/out, bf16 WMMA internally.
// Block: 256 threads = 8 waves; each wave computes a 16-row stripe x all N.
// W^T staged in LDS in <=64KB chunks (128 cols x K=256 x bf16).
template <int K, int N>
__global__ __launch_bounds__(256) void gcn_gemm_wmma_bf16(
    const float* __restrict__ A, const float* __restrict__ Wg,
    float* __restrict__ C) {
  constexpr int KB  = K / 32;                 // K blocks per WMMA chain
  constexpr int NCH = (N < 128) ? N : 128;    // cols per LDS chunk (<=64KB)
  __shared__ __attribute__((aligned(16))) __bf16 sWT[NCH * K];  // transposed

  const int tid  = threadIdx.x;
  const int lane = tid & 31;
  const int wave = tid >> 5;          // 0..7
  const int kh   = lane >> 4;         // K half-block select (ISA A/B layout)
  const int ln   = lane & 15;         // M row (A) / N col (B,C,D)
  const size_t rowBase = (size_t)blockIdx.x * 128;
  const int    rowW    = wave * 16;

  // A fragments for all K, per ISA 16-bit A layout:
  //   elem e<8 : k = kb*32 + 8*kh + e ; elem e>=8 : k = kb*32 + 16 + 8*kh + (e-8)
  v16bf afrag[KB];
  {
    const float* arow = A + (rowBase + rowW + ln) * (size_t)K;
    #pragma unroll
    for (int kb = 0; kb < KB; ++kb) {
      const float4 l0 = *(const float4*)(arow + kb * 32 + kh * 8);
      const float4 l1 = *(const float4*)(arow + kb * 32 + kh * 8 + 4);
      const float4 h0 = *(const float4*)(arow + kb * 32 + 16 + kh * 8);
      const float4 h1 = *(const float4*)(arow + kb * 32 + 16 + kh * 8 + 4);
      v16bf t;
      t[0]=(__bf16)l0.x;  t[1]=(__bf16)l0.y;  t[2]=(__bf16)l0.z;  t[3]=(__bf16)l0.w;
      t[4]=(__bf16)l1.x;  t[5]=(__bf16)l1.y;  t[6]=(__bf16)l1.z;  t[7]=(__bf16)l1.w;
      t[8]=(__bf16)h0.x;  t[9]=(__bf16)h0.y;  t[10]=(__bf16)h0.z; t[11]=(__bf16)h0.w;
      t[12]=(__bf16)h1.x; t[13]=(__bf16)h1.y; t[14]=(__bf16)h1.z; t[15]=(__bf16)h1.w;
      afrag[kb] = t;
    }
  }

  for (int nc = 0; nc < N; nc += NCH) {
    __syncthreads();
    // stage W^T chunk (coalesced global reads, fp32 -> bf16)
    for (int idx = tid; idx < K * NCH; idx += 256) {
      int k = idx / NCH;
      int n = idx - k * NCH;
      sWT[n * K + k] = (__bf16)Wg[(size_t)k * N + nc + n];
    }
    __syncthreads();

    #pragma unroll
    for (int nt = 0; nt < NCH / 16; ++nt) {
      v8f acc = {};
      // B layout: lane ln = column n; elem e<8: k = 8*kh + e, e>=8: k = 16 + 8*kh + (e-8)
      const __bf16* wbase = &sWT[(nt * 16 + ln) * K];
      #pragma unroll
      for (int kb = 0; kb < KB; ++kb) {
        v8bf blo = *(const v8bf*)(wbase + kb * 32 + kh * 8);
        v8bf bhi = *(const v8bf*)(wbase + kb * 32 + 16 + kh * 8);
        v16bf b;
        #pragma unroll
        for (int e = 0; e < 8; ++e) { b[e] = blo[e]; b[8 + e] = bhi[e]; }
        acc = __builtin_amdgcn_wmma_f32_16x16x32_bf16(
            /*neg_a=*/false, afrag[kb], /*neg_b=*/false, b,
            /*c_mod=*/(short)0, acc, /*reuse_a=*/false, /*reuse_b=*/false);
      }
      // C/D layout: VGPR r -> M = r + 8*kh, N = ln
      const int n0 = nc + nt * 16 + ln;
      float* crow = C + (rowBase + rowW + 8 * kh) * (size_t)N + n0;
      #pragma unroll
      for (int r = 0; r < 8; ++r) crow[(size_t)r * N] = acc[r];
    }
  }
}

// --------------------------- edge aggregation -------------------------------
// out[i,:] starts with the self-loop term dis[i]^2 * h[i,:]
__global__ void gcn_agg_init(const float* __restrict__ h,
                             const float* __restrict__ dis,
                             float* __restrict__ out, int total, int shift) {
  int g = GID; if (g >= total) return;
  int i = g >> shift;
  float s = dis[i];
  out[g] = s * s * h[g];
}
// scatter: out[dst,:] += dis[src]*dis[dst] * h[src,:]
__global__ void gcn_agg_edge(const float* __restrict__ h,
                             const float* __restrict__ dis,
                             const int* __restrict__ src,
                             const int* __restrict__ dst,
                             float* __restrict__ out, int E, int shift) {
  int g = GID;
  int e = g >> shift;
  if (e >= E) return;
  int d = g & ((1 << shift) - 1);
  int s = src[e], t = dst[e];
  float nrm = dis[s] * dis[t];
  unsafeAtomicAdd(&out[((size_t)t << shift) + d],
                  nrm * h[((size_t)s << shift) + d]);
}
__global__ void gcn_agg_fin(float* __restrict__ out, const float* __restrict__ bias,
                            int total, int shift, int relu) {
  int g = GID; if (g >= total) return;
  int d = g & ((1 << shift) - 1);
  float v = out[g] + bias[d];
  out[g] = relu ? fmaxf(v, 0.0f) : v;
}

// --------------------------- pooling + pairs --------------------------------
// batch = repeat(arange(NIMG), 128) -> each image owns 128 contiguous nodes.
__global__ void gcn_pool(const float* __restrict__ h, float* __restrict__ img) {
  int g = blockIdx.x;          // image
  int d = threadIdx.x;         // 0..127 (OUT_DIM)
  const float* base = h + (size_t)g * 128 * 128;
  float s = 0.0f;
  #pragma unroll 4
  for (int j = 0; j < 128; ++j) s += base[j * 128 + d];
  img[g * 128 + d] = s * (1.0f / 128.0f);
}
// one wave32 per pair: dot([img[a], img[b]], fcW) -> sigmoid
__global__ void gcn_pairs(const float* __restrict__ img, const int* __restrict__ pairs,
                          const float* __restrict__ fcW, const float* __restrict__ fcb,
                          float* __restrict__ out, int np) {
  int g = GID;
  int p = g >> 5, lane = g & 31;
  if (p >= np) return;
  int a = pairs[2 * p], b = pairs[2 * p + 1];
  float s = 0.0f;
  for (int j = lane; j < 128; j += 32)
    s += img[a * 128 + j] * fcW[j] + img[b * 128 + j] * fcW[128 + j];
  #pragma unroll
  for (int off = 16; off > 0; off >>= 1) s += __shfl_xor(s, off, 32);
  if (lane == 0) out[p] = 1.0f / (1.0f + expf(-(s + fcb[0])));
}

// ------------------------------- launch -------------------------------------
extern "C" void kernel_launch(void* const* d_in, const int* in_sizes, int n_in,
                              void* d_out, int out_size, void* d_ws, size_t ws_size,
                              hipStream_t stream) {
  const float* x    = (const float*)d_in[0];
  const int*   ei   = (const int*)  d_in[1];   // [2, E] flat
  const int*   prs  = (const int*)  d_in[3];   // [NP, 2] flat
  const float* W1   = (const float*)d_in[4];
  const float* b1   = (const float*)d_in[5];
  const float* W2   = (const float*)d_in[6];
  const float* b2   = (const float*)d_in[7];
  const float* fcW  = (const float*)d_in[8];
  const float* fcb  = (const float*)d_in[9];
  float*       out  = (float*)d_out;

  const int N    = in_sizes[2];        // 128000 nodes
  const int E    = in_sizes[1] / 2;    // 384000 edges
  const int NP   = in_sizes[3] / 2;    // 4096 pairs
  const int NIMG = N / 128;            // 1000 images
  const int* src = ei;
  const int* dst = ei + E;

  // workspace layout (512B aligned): dis | h1 | a1 | img ; h2 reuses h1,
  // a2 reuses a1 (lifetimes are disjoint on the serialized stream).
  size_t off = 0;
  auto take = [&](size_t bytes) { size_t o = off; off = (off + bytes + 511) & ~(size_t)511; return o; };
  char* w = (char*)d_ws;
  float* dis = (float*)(w + take((size_t)N * 4));
  float* h1  = (float*)(w + take((size_t)N * 256 * 4));
  float* a1  = (float*)(w + take((size_t)N * 256 * 4));
  float* img = (float*)(w + take((size_t)NIMG * 128 * 4));
  float* h2  = h1;   // N*128 fits in h1 region
  float* a2  = a1;   // N*128 fits in a1 region

  const int T = 256;
  // degrees -> deg^{-1/2}  (deg includes self loop, so >= 1)
  gcn_fill_one<<<(N + T - 1) / T, T, 0, stream>>>(dis, N);
  gcn_deg_edge<<<(E + T - 1) / T, T, 0, stream>>>(dst, dis, E);
  gcn_deg_fin <<<(N + T - 1) / T, T, 0, stream>>>(dis, N);

  // layer 1: h1 = x @ W1 ; a1 = scatter(norm*h1) + b1 ; relu
  gcn_gemm_wmma_bf16<256, 256><<<N / 128, 256, 0, stream>>>(x, W1, h1);
  gcn_agg_init<<<(N * 256) / T, T, 0, stream>>>(h1, dis, a1, N * 256, 8);
  gcn_agg_edge<<<E, T, 0, stream>>>(h1, dis, src, dst, a1, E, 8);
  gcn_agg_fin <<<(N * 256) / T, T, 0, stream>>>(a1, b1, N * 256, 8, 1);

  // layer 2: h2 = a1 @ W2 ; a2 = scatter(norm*h2) + b2
  gcn_gemm_wmma_bf16<256, 128><<<N / 128, 256, 0, stream>>>(a1, W2, h2);
  gcn_agg_init<<<(N * 128) / T, T, 0, stream>>>(h2, dis, a2, N * 128, 7);
  gcn_agg_edge<<<(E * 128) / T, T, 0, stream>>>(h2, dis, src, dst, a2, E, 7);
  gcn_agg_fin <<<(N * 128) / T, T, 0, stream>>>(a2, b2, N * 128, 7, 0);

  // mean pool per image, then pair FC + sigmoid
  gcn_pool <<<NIMG, 128, 0, stream>>>(a2, img);
  gcn_pairs<<<(NP * 32 + T - 1) / T, T, 0, stream>>>(img, prs, fcW, fcb, out, NP);
}
